// GATNet_47811575939383
// MI455X (gfx1250) — matrix operand order repaired and checked by reading, not compiled, not run on hardware
//
#include <hip/hip_runtime.h>
#include <hip/hip_bf16.h>
#include <stdint.h>

// Problem constants (match reference)
static constexpr int kN  = 4096;
static constexpr int kDQ = 128;
static constexpr int kK  = 3;
static constexpr int BJ  = 64;    // j-chunk per stage (2 WMMA K-steps)
static constexpr int TS  = 68;    // trans LDS tile row stride in floats (pad: conflict-free)

typedef __attribute__((ext_vector_type(16))) _Float16 v16h;
typedef __attribute__((ext_vector_type(8)))  float    v8f;

// ---------------------------------------------------------------------------
// Encoder: h = concat(x, comms) @ W_enc + b_enc     (4096 x 96) @ (96 x 128)
// ---------------------------------------------------------------------------
__global__ __launch_bounds__(128) void enc_kernel(const float* __restrict__ x,
                                                  const float* __restrict__ comms,
                                                  const float* __restrict__ W_enc,
                                                  const float* __restrict__ b_enc,
                                                  float* __restrict__ h) {
  const int row = blockIdx.x;
  const int c   = threadIdx.x;
  __shared__ float xr[96];
  if (c < 64)       xr[c] = x[row * 64 + c];
  else if (c < 96)  xr[c] = comms[row * 32 + (c - 64)];
  __syncthreads();
  float acc = b_enc[c];
  #pragma unroll 8
  for (int d = 0; d < 96; ++d) acc += xr[d] * W_enc[d * kDQ + c];
  h[(size_t)row * kDQ + c] = acc;
}

// ---------------------------------------------------------------------------
// wq[d] = sum_q W_ad[k][d,q] * wa[q],  cq = b_ad[k] . wa
// (row-constant score terms cancel inside the row softmax)
// ---------------------------------------------------------------------------
__global__ __launch_bounds__(128) void wq_kernel(const float* __restrict__ W_ad_k,
                                                 const float* __restrict__ wa,
                                                 const float* __restrict__ b_ad_k,
                                                 float* __restrict__ wq,
                                                 float* __restrict__ cq) {
  const int d = threadIdx.x;
  float acc = 0.f;
  #pragma unroll 8
  for (int q = 0; q < kDQ; ++q) acc += W_ad_k[d * kDQ + q] * wa[q];
  wq[d] = acc;
  __shared__ float red[128];
  red[d] = b_ad_k[d] * wa[d];
  __syncthreads();
  for (int s = 64; s > 0; s >>= 1) {
    if (d < s) red[d] += red[d + s];
    __syncthreads();
  }
  if (d == 0) *cq = red[0];
}

// ---------------------------------------------------------------------------
// Per-node projections: q[j] = h_j . wq + cq ;  vT[c, j] = (h_j @ W_av + b_av)[c]
// V is written TRANSPOSED in f16 so attention B-fragments are contiguous rows.
// ---------------------------------------------------------------------------
__global__ __launch_bounds__(128) void qv_kernel(const float* __restrict__ h,
                                                 const float* __restrict__ W_av_k,
                                                 const float* __restrict__ b_av_k,
                                                 const float* __restrict__ wq,
                                                 const float* __restrict__ cq,
                                                 float* __restrict__ qout,
                                                 _Float16* __restrict__ vT) {
  const int row = blockIdx.x;
  const int c   = threadIdx.x;
  __shared__ float hr[128];
  __shared__ float red[128];
  hr[c] = h[(size_t)row * kDQ + c];
  __syncthreads();
  float acc = b_av_k[c];
  #pragma unroll 8
  for (int d = 0; d < kDQ; ++d) acc += hr[d] * W_av_k[d * kDQ + c];
  vT[(size_t)c * kN + row] = (_Float16)acc;           // transposed store
  red[c] = hr[c] * wq[c];
  __syncthreads();
  for (int s = 64; s > 0; s >>= 1) {
    if (c < s) red[c] += red[c + s];
    __syncthreads();
  }
  if (c == 0) qout[row] = red[0] + *cq;
}

// ---------------------------------------------------------------------------
// Static softmax shift: M = max(q) + 1  (trans <= 1 so s <= M everywhere);
// rewrite q <- q - M in place. Softmax is shift-invariant, so this is exact.
// ---------------------------------------------------------------------------
__global__ __launch_bounds__(256) void qnorm_kernel(float* __restrict__ q) {
  __shared__ float red[256];
  const int t = threadIdx.x;
  float m = -__builtin_inff();
  for (int j = t; j < kN; j += 256) m = fmaxf(m, q[j]);
  red[t] = m;
  __syncthreads();
  for (int s = 128; s > 0; s >>= 1) {
    if (t < s) red[t] = fmaxf(red[t], red[t + s]);
    __syncthreads();
  }
  const float M = red[0] + 1.0f;
  for (int j = t; j < kN; j += 256) q[j] -= M;
}

// ---------------------------------------------------------------------------
// Fused attention round with static-shift softmax:
//   h_out[i,:] = (1/L_i) * sum_j exp(qm[j] + trans[j,i]) * v[j,:]
// 4 waves/block, 16 rows/wave. trans tile (64j x 64i) staged to LDS with
// global_load_async_to_lds_b128; P@V on v_wmma_f32_16x16x32_f16 with B
// fragments streamed straight from the transposed f16 V in global/L2.
// Softmax denominators come from an extra WMMA against a ones-B tile:
// D[m,n] = sum_k P[m,k] is replicated into every column, so every lane holds
// its own rows' sums with zero cross-lane traffic.
// ---------------------------------------------------------------------------
__global__ __launch_bounds__(128) void attn_kernel(const float* __restrict__ qm,
                                                   const _Float16* __restrict__ vT,
                                                   const float* __restrict__ trans,
                                                   float* __restrict__ hout) {
  __shared__ float ldsT[BJ * TS];   // 64 x 68 floats = 17408 B

  const int tid  = threadIdx.x;
  const int wv   = tid >> 5;
  const int lane = tid & 31;
  const int half = lane >> 4;       // WMMA A/B K-half
  const int tm   = lane & 15;       // A row / C column within tile
  const int iblk = blockIdx.x * 64;
  const int i0   = iblk + wv * 16;

  v16h bones;                       // ones B-fragment: row-sum engine
  #pragma unroll
  for (int e = 0; e < 16; ++e) bones[e] = (_Float16)1.0f;

  v8f acc[8];                       // 16 rows x 128 cols of un-normalized output
  v8f accS;                         // row sums (replicated across columns)
  #pragma unroll
  for (int nt = 0; nt < 8; ++nt)
    #pragma unroll
    for (int r = 0; r < 8; ++r) acc[nt][r] = 0.f;
  #pragma unroll
  for (int r = 0; r < 8; ++r) accS[r] = 0.f;

  for (int j0 = 0; j0 < kN; j0 += BJ) {
    __syncthreads();   // previous tile fully consumed before overwrite

    // ---- async-copy trans[j0..j0+63, iblk..iblk+63] into padded LDS tile
    #pragma unroll
    for (int t = 0; t < 8; ++t) {
      const int idx = tid + t * 128;            // 0..1023 16B-transfers
      const int r   = idx >> 4;                 // tile row (j - j0)
      const int c4  = (idx & 15) << 2;          // tile col (float index)
      const float*   gp = &trans[(size_t)(j0 + r) * kN + iblk + c4];
      const uint32_t lp = (uint32_t)(uintptr_t)(&ldsT[r * TS + c4]);
      asm volatile("global_load_async_to_lds_b128 %0, %1, off"
                   :: "v"(lp), "v"(gp) : "memory");
    }
    asm volatile("s_wait_asynccnt 0x0" ::: "memory");
    __syncthreads();

    // ---- two WMMA K-steps of 32 j's each
    #pragma unroll
    for (int kc = 0; kc < 2; ++kc) {
      // P fragment (A layout): lane (tm, half) holds row tm, its 16 K-slots
      v16h a;
      #pragma unroll
      for (int e = 0; e < 16; ++e) {
        const int kk = kc * 32 + e + 8 * half + ((e & 8) ? 8 : 0);
        const float tv = ldsT[kk * TS + wv * 16 + tm];
        const float p  = (tv == 0.f) ? 0.f : __expf(qm[j0 + kk] + tv);
        a[e] = (_Float16)p;
      }

      // Row sums via the matrix pipe: accS += P @ ones
      accS = __builtin_amdgcn_wmma_f32_16x16x32_f16(
          false, a, false, bones, (short)0, accS, false, false);

      // O += P @ V over the 8 column tiles; B fragment is two 16B chunks
      // (k = 8*half.. and 16+8*half..) of the transposed V row for this lane's column.
      #pragma unroll
      for (int nt = 0; nt < 8; ++nt) {
        const int col = nt * 16 + tm;
        typedef __attribute__((ext_vector_type(8))) _Float16 v8h_;
        const _Float16* bp = &vT[(size_t)col * kN + j0 + kc * 32 + half * 8];
        const v8h_ b0 = *(const v8h_*)(bp);
        const v8h_ b1 = *(const v8h_*)(bp + 16);
        const v16h b = __builtin_shufflevector(b0, b1, 0,1,2,3,4,5,6,7,8,9,10,11,12,13,14,15);
        acc[nt] = __builtin_amdgcn_wmma_f32_16x16x32_f16(
            /*neg_a=*/false, a, /*neg_b=*/false, b,
            /*c_mod=*/(short)0, acc[nt], /*reuse_a=*/false, /*reuse_b=*/false);
      }
    }
  }

  // ---- normalize and write h_out (C layout: lane owns col tm of rows r+8*half)
  float invL[8];
  #pragma unroll
  for (int r = 0; r < 8; ++r) invL[r] = 1.0f / accS[r];

  #pragma unroll
  for (int nt = 0; nt < 8; ++nt) {
    const int col = nt * 16 + tm;
    #pragma unroll
    for (int r = 0; r < 8; ++r) {
      const int row = i0 + r + 8 * half;
      hout[(size_t)row * kDQ + col] = acc[nt][r] * invL[r];
    }
  }
}

// ---------------------------------------------------------------------------
// Decoder: out[i] = h_i . W_dec + b_dec  (+ -inf where mask==0)
// ---------------------------------------------------------------------------
__global__ __launch_bounds__(256) void dec_kernel(const float* __restrict__ h,
                                                  const float* __restrict__ W_dec,
                                                  const float* __restrict__ b_dec,
                                                  const int* __restrict__ mask,
                                                  float* __restrict__ out) {
  const int i = blockIdx.x * blockDim.x + threadIdx.x;
  if (i >= kN) return;
  float acc = *b_dec;
  #pragma unroll 8
  for (int d = 0; d < kDQ; ++d) acc += h[(size_t)i * kDQ + d] * W_dec[d];
  out[i] = acc + (mask[i] == 0 ? -__builtin_inff() : 0.f);
}

// ---------------------------------------------------------------------------
extern "C" void kernel_launch(void* const* d_in, const int* in_sizes, int n_in,
                              void* d_out, int out_size, void* d_ws, size_t ws_size,
                              hipStream_t stream) {
  const float* x      = (const float*)d_in[0];
  const float* comms  = (const float*)d_in[1];
  const float* trans  = (const float*)d_in[2];
  const int*   mask   = (const int*)d_in[3];
  const float* W_enc  = (const float*)d_in[4];
  const float* b_enc  = (const float*)d_in[5];
  const float* W_ad   = (const float*)d_in[6];
  const float* b_ad   = (const float*)d_in[7];
  const float* w_att  = (const float*)d_in[8];
  // d_in[9] = b_att: cancels inside the row softmax, unused.
  const float* W_av   = (const float*)d_in[10];
  const float* b_av   = (const float*)d_in[11];
  const float* W_dec  = (const float*)d_in[12];
  const float* b_dec  = (const float*)d_in[13];
  float* out = (float*)d_out;

  char* ws = (char*)d_ws;
  const size_t hBytes = (size_t)kN * kDQ * sizeof(float);        // 2 MB each
  float*    hA   = (float*)(ws);
  float*    hB   = (float*)(ws + hBytes);
  _Float16* vT   = (_Float16*)(ws + 2 * hBytes);                 // 1 MB, transposed V
  float*    qbuf = (float*)(ws + 2 * hBytes + (size_t)kN * kDQ * sizeof(_Float16));
  float*    wq   = qbuf + kN;
  float*    cq   = wq + kDQ;

  enc_kernel<<<kN, 128, 0, stream>>>(x, comms, W_enc, b_enc, hA);

  float* hcur = hA;
  float* hnxt = hB;
  for (int k = 0; k < kK; ++k) {
    wq_kernel<<<1, 128, 0, stream>>>(W_ad + (size_t)k * kDQ * kDQ,
                                     w_att + (size_t)k * 2 * kDQ,
                                     b_ad + (size_t)k * kDQ, wq, cq);
    qv_kernel<<<kN, 128, 0, stream>>>(hcur, W_av + (size_t)k * kDQ * kDQ,
                                      b_av + (size_t)k * kDQ, wq, cq, qbuf, vT);
    qnorm_kernel<<<1, 256, 0, stream>>>(qbuf);
    attn_kernel<<<kN / 64, 128, 0, stream>>>(qbuf, vT, trans, hnxt);
    float* t = hcur; hcur = hnxt; hnxt = t;
  }

  dec_kernel<<<(kN + 255) / 256, 256, 0, stream>>>(hcur, W_dec, b_dec, mask, out);
}
